// DGPT_47614007443573
// MI455X (gfx1250) — matrix operand order repaired and checked by reading, not compiled, not run on hardware
//
#include <hip/hip_runtime.h>
#include <hip/hip_bf16.h>
#include <math.h>

typedef __attribute__((ext_vector_type(16))) __bf16 v16bf;
typedef __attribute__((ext_vector_type(8)))  float  v8f;

namespace cfg {
constexpr int B = 8, T = 1024, E = 768, H = 12, D = 64, L = 12, V = 50304;
}

// ---------------------------------------------------------------- converts
__global__ __launch_bounds__(256) void cvt_kernel(const float* __restrict__ in,
                                                  __bf16* __restrict__ out, size_t n) {
  size_t i = (size_t)blockIdx.x * 256 + threadIdx.x;
  if (i < n) out[i] = (__bf16)in[i];
}

// ---------------------------------------------------------------- embedding
__global__ __launch_bounds__(256) void embed_kernel(const int* __restrict__ idx,
                                                    const float* __restrict__ wte,
                                                    const float* __restrict__ wpe,
                                                    float* __restrict__ x) {
  using namespace cfg;
  size_t i = (size_t)blockIdx.x * 256 + threadIdx.x;
  if (i >= (size_t)B * T * E) return;
  int e = (int)(i % E);
  size_t tok = i / E;
  int t = (int)(tok % T);
  x[i] = wte[(size_t)idx[tok] * E + e] + wpe[(size_t)t * E + e];
}

// ---------------------------------------------------------------- layernorm (fp32 in, bf16 out)
__global__ __launch_bounds__(256) void ln_kernel(const float* __restrict__ x,
                                                 const float* __restrict__ w,
                                                 const float* __restrict__ bsrc,
                                                 __bf16* __restrict__ out, int ntok) {
  using namespace cfg;
  int wave = threadIdx.x >> 5, lane = threadIdx.x & 31;
  int tok = blockIdx.x * 8 + wave;
  if (tok >= ntok) return;
  const float* row = x + (size_t)tok * E;
  float vals[E / 32];
  float s = 0.f;
#pragma unroll
  for (int i = 0; i < E / 32; ++i) { vals[i] = row[lane + i * 32]; s += vals[i]; }
#pragma unroll
  for (int m = 1; m < 32; m <<= 1) s += __shfl_xor(s, m, 32);
  float mean = s * (1.0f / E);
  float v = 0.f;
#pragma unroll
  for (int i = 0; i < E / 32; ++i) { float d = vals[i] - mean; v += d * d; }
#pragma unroll
  for (int m = 1; m < 32; m <<= 1) v += __shfl_xor(v, m, 32);
  float rstd = rsqrtf(v * (1.0f / E) + 1e-5f);
  __bf16* orow = out + (size_t)tok * E;
#pragma unroll
  for (int i = 0; i < E / 32; ++i) {
    int c = lane + i * 32;
    orow[c] = (__bf16)((vals[i] - mean) * rstd * w[c] + bsrc[c]);
  }
}

// ---------------------------------------------------------------- gather last-token rows
__global__ __launch_bounds__(256) void gather_kernel(const __bf16* __restrict__ lnbuf,
                                                     __bf16* __restrict__ x8) {
  using namespace cfg;
  int i = blockIdx.x * 256 + threadIdx.x;
  if (i >= B * E) return;
  int b = i / E, e = i % E;
  x8[i] = lnbuf[(size_t)(b * T + (T - 1)) * E + e];
}

// ---------------------------------------------------------------- WMMA GEMM
// C[M,N] = epilogue(A[M,K](bf16) @ W(bf16) + bias [+ residual])
// EPI: 0 = bf16 out, 1 = bf16 out + exact GELU, 2 = f32 out + residual add
// TRANSB: W stored [N,K] row-major (for x @ wte^T)
template <int EPI, bool TRANSB>
__global__ __launch_bounds__(256) void gemm_kernel(const __bf16* __restrict__ A,
                                                   const __bf16* __restrict__ Wm,
                                                   const float* __restrict__ bias,
                                                   const float* __restrict__ res,
                                                   void* __restrict__ Cout,
                                                   int M, int N, int K) {
  __shared__ __bf16 lA[128][40];   // 128x32 tile (+pad: 80B row stride, 16B aligned)
  __shared__ __bf16 lBt[128][40];  // [col][k] transposed tile
  const int tid = threadIdx.x;
  const int lane = tid & 31, wave = tid >> 5;
  const int wm = wave & 3, wn = wave >> 2;  // 4x2 wave grid, 32x64 per wave
  const int lh = lane >> 4, lr = lane & 15;
  const int m0 = blockIdx.y * 128;
  const int n0 = blockIdx.x * 128;

  const int arow = tid >> 1, acol = (tid & 1) * 16;   // A staging
  const int brow = tid >> 3, bcol = (tid & 7) * 16;   // B staging (normal)
  const int tcol = tid & 127, tkb = (tid >> 7) * 16;  // B staging (transposed)

  v8f acc[2][4];
  const v8f vzero = {0.f, 0.f, 0.f, 0.f, 0.f, 0.f, 0.f, 0.f};
#pragma unroll
  for (int i = 0; i < 2; ++i)
#pragma unroll
    for (int j = 0; j < 4; ++j) acc[i][j] = vzero;

  uint4 ra0, ra1, rb0, rb1;
  auto load_tiles = [&](int k0) {
    uint4 z = {0, 0, 0, 0};
    ra0 = z; ra1 = z;
    if (m0 + arow < M) {
      const uint4* src = reinterpret_cast<const uint4*>(A + (size_t)(m0 + arow) * K + k0 + acol);
      ra0 = src[0]; ra1 = src[1];
    }
    if constexpr (!TRANSB) {
      const uint4* src = reinterpret_cast<const uint4*>(Wm + (size_t)(k0 + brow) * N + n0 + bcol);
      rb0 = src[0]; rb1 = src[1];
    } else {
      const uint4* src = reinterpret_cast<const uint4*>(Wm + (size_t)(n0 + tcol) * K + k0 + tkb);
      rb0 = src[0]; rb1 = src[1];
    }
  };
  auto store_tiles = [&]() {
    *reinterpret_cast<uint4*>(&lA[arow][acol]) = ra0;
    *reinterpret_cast<uint4*>(&lA[arow][acol + 8]) = ra1;
    if constexpr (!TRANSB) {
      union { uint4 u[2]; __bf16 h[16]; } t;
      t.u[0] = rb0; t.u[1] = rb1;
#pragma unroll
      for (int i = 0; i < 16; ++i) lBt[bcol + i][brow] = t.h[i];
    } else {
      *reinterpret_cast<uint4*>(&lBt[tcol][tkb]) = rb0;
      *reinterpret_cast<uint4*>(&lBt[tcol][tkb + 8]) = rb1;
    }
  };

  load_tiles(0);
  for (int k0 = 0; k0 < K; k0 += 32) {
    store_tiles();
    __syncthreads();
    if (k0 + 32 < K) load_tiles(k0 + 32);  // overlap next global load with WMMAs
    if (k0 + 64 < K) {                     // far prefetch -> global_prefetch_b8
      __builtin_prefetch(A + (size_t)(m0 + arow) * K + k0 + 64 + acol, 0, 0);
      if constexpr (!TRANSB)
        __builtin_prefetch(Wm + (size_t)(k0 + 64 + brow) * N + n0 + bcol, 0, 0);
      else
        __builtin_prefetch(Wm + (size_t)(n0 + tcol) * K + k0 + 64 + tkb, 0, 0);
    }

    v16bf af[2], bfr[4];
#pragma unroll
    for (int i = 0; i < 2; ++i) {
      int row = wm * 32 + i * 16 + lr;
      union { uint4 u[2]; v16bf v; } fa;
      fa.u[0] = *reinterpret_cast<const uint4*>(&lA[row][lh * 8]);
      fa.u[1] = *reinterpret_cast<const uint4*>(&lA[row][16 + lh * 8]);
      af[i] = fa.v;
    }
#pragma unroll
    for (int j = 0; j < 4; ++j) {
      int col = wn * 64 + j * 16 + lr;
      union { uint4 u[2]; v16bf v; } fb;
      fb.u[0] = *reinterpret_cast<const uint4*>(&lBt[col][lh * 8]);
      fb.u[1] = *reinterpret_cast<const uint4*>(&lBt[col][16 + lh * 8]);
      bfr[j] = fb.v;
    }
#pragma unroll
    for (int i = 0; i < 2; ++i)
#pragma unroll
      for (int j = 0; j < 4; ++j)
        acc[i][j] = __builtin_amdgcn_wmma_f32_16x16x32_bf16(
            false, af[i], false, bfr[j], (short)0, acc[i][j], false, false);
    __syncthreads();
  }

#pragma unroll
  for (int i = 0; i < 2; ++i) {
#pragma unroll
    for (int j = 0; j < 4; ++j) {
      int col = n0 + wn * 64 + j * 16 + lr;
      float bv = bias ? bias[col] : 0.0f;
#pragma unroll
      for (int r = 0; r < 8; ++r) {
        int row = m0 + wm * 32 + i * 16 + r + 8 * lh;
        if (row < M) {
          float v = acc[i][j][r] + bv;
          if constexpr (EPI == 2) {
            float rr = res ? res[(size_t)row * N + col] : 0.0f;
            ((float*)Cout)[(size_t)row * N + col] = rr + v;
          } else {
            if constexpr (EPI == 1) v = 0.5f * v * (1.0f + erff(v * 0.70710678118654752f));
            ((__bf16*)Cout)[(size_t)row * N + col] = (__bf16)v;
          }
        }
      }
    }
  }
}

// ---------------------------------------------------------------- fused causal flash attention
// One wave per (b, h, 32-query tile): two 16-row Q tiles share K frags + LDS V tile.
// QK^T and P*V via WMMA bf16; row-sum via ones-matrix WMMA; online softmax fp32.
// 1/sqrt(D) is pre-folded into the Q fragments (exact: power of two in bf16).
template <bool MASKED>
__device__ __forceinline__ void attn_step(const __bf16* __restrict__ qkv, int b, int h,
                                          int q0, int kk, int lane, int lh, int lr,
                                          const v16bf (&qa)[2][2], const v16bf& ones,
                                          v8f (&o)[2][4], float (&m)[2][8],
                                          float (&lsum)[2][8],
                                          __bf16 (*lP)[40], __bf16 (*lV)[72]) {
  using namespace cfg;
  const v8f vzero = {0.f, 0.f, 0.f, 0.f, 0.f, 0.f, 0.f, 0.f};

  // stage V tile (32 keys x 64 d): lane owns key row kk+lane, all-b128 traffic
  {
    const uint4* vsrc = reinterpret_cast<const uint4*>(
        qkv + (size_t)(b * T + kk + lane) * (3 * E) + 2 * E + h * D);
    uint4* vdst = reinterpret_cast<uint4*>(&lV[lane][0]);
#pragma unroll
    for (int i = 0; i < 8; ++i) vdst[i] = vsrc[i];
  }

  // S = Q @ K^T : K col-tile frags (contiguous 32B global loads) shared by both row tiles
  v8f s[2][2];
#pragma unroll
  for (int ct = 0; ct < 2; ++ct) {
    int key = kk + ct * 16 + lr;
    const __bf16* kbase = qkv + (size_t)(b * T + key) * (3 * E) + E + h * D;
    union { uint4 u[2]; v16bf v; } k0u, k1u;
    k0u.u[0] = reinterpret_cast<const uint4*>(kbase + lh * 16)[0];
    k0u.u[1] = reinterpret_cast<const uint4*>(kbase + lh * 16)[1];
    k1u.u[0] = reinterpret_cast<const uint4*>(kbase + 32 + lh * 16)[0];
    k1u.u[1] = reinterpret_cast<const uint4*>(kbase + 32 + lh * 16)[1];
#pragma unroll
    for (int rt = 0; rt < 2; ++rt) {
      v8f st = vzero;
      st = __builtin_amdgcn_wmma_f32_16x16x32_bf16(false, qa[rt][0], false, k0u.v, (short)0, st, false, false);
      st = __builtin_amdgcn_wmma_f32_16x16x32_bf16(false, qa[rt][1], false, k1u.v, (short)0, st, false, false);
      s[rt][ct] = st;
    }
  }

  // online softmax (row = rt*16 + r + 8*lh); only running-max needs lane shuffles
#pragma unroll
  for (int rt = 0; rt < 2; ++rt) {
#pragma unroll
    for (int r = 0; r < 8; ++r) {
      float v0 = s[rt][0][r];
      float v1 = s[rt][1][r];
      if constexpr (MASKED) {
        int q = q0 + rt * 16 + r + 8 * lh;
        if (kk + lr > q) v0 = -INFINITY;
        if (kk + 16 + lr > q) v1 = -INFINITY;
      }
      float rm = fmaxf(v0, v1);
#pragma unroll
      for (int msk = 1; msk < 16; msk <<= 1) rm = fmaxf(rm, __shfl_xor(rm, msk, 32));
      float mn = fmaxf(m[rt][r], rm);  // finite: key 0 is visible to every row
      float alpha = __expf(m[rt][r] - mn);
      m[rt][r] = mn;
      lsum[rt][r] *= alpha;
#pragma unroll
      for (int j = 0; j < 4; ++j) o[rt][j][r] *= alpha;
      lP[rt * 16 + r + 8 * lh][lr] = (__bf16)__expf(v0 - mn);
      lP[rt * 16 + r + 8 * lh][16 + lr] = (__bf16)__expf(v1 - mn);
    }
  }
  __syncthreads();

  // V fragments from LDS: base-reg + immediate-offset DS reads (no VALU addr math)
  v16bf vb[4];
#pragma unroll
  for (int j = 0; j < 4; ++j)
#pragma unroll
    for (int e = 0; e < 16; ++e) vb[j][e] = lV[(e & 7) + 8 * lh + 16 * (e >> 3)][j * 16 + lr];

#pragma unroll
  for (int rt = 0; rt < 2; ++rt) {
    union { uint4 u[2]; v16bf v; } fp;
    fp.u[0] = *reinterpret_cast<const uint4*>(&lP[rt * 16 + lr][lh * 8]);
    fp.u[1] = *reinterpret_cast<const uint4*>(&lP[rt * 16 + lr][16 + lh * 8]);
    v16bf pa = fp.v;
    // row-sum on the matrix pipe: rowsum(P) = P @ ones (replaces 32 ds_bpermutes)
    v8f rs = vzero;
    rs = __builtin_amdgcn_wmma_f32_16x16x32_bf16(false, pa, false, ones, (short)0, rs, false, false);
#pragma unroll
    for (int r = 0; r < 8; ++r) lsum[rt][r] += rs[r];
#pragma unroll
    for (int j = 0; j < 4; ++j)
      o[rt][j] = __builtin_amdgcn_wmma_f32_16x16x32_bf16(false, pa, false, vb[j], (short)0, o[rt][j], false, false);
  }
  __syncthreads();
}

__global__ __launch_bounds__(32) void attn_kernel(const __bf16* __restrict__ qkv,
                                                  __bf16* __restrict__ y) {
  using namespace cfg;
  const int lane = threadIdx.x;
  const int lh = lane >> 4, lr = lane & 15;
  const int QT = T / 32;
  int bid = blockIdx.x;
  int qt = bid % QT;
  int h = (bid / QT) % H;
  int b = bid / (QT * H);
  int q0 = qt * 32;

  __shared__ __bf16 lP[32][40];
  __shared__ __bf16 lV[32][72];

  v16bf ones;
#pragma unroll
  for (int e = 0; e < 16; ++e) ones[e] = (__bf16)1.0f;

  // Q fragments (A layout): contiguous 16B global loads, then pre-scale by 1/sqrt(D)
  // once per kernel (exact power-of-two scale in bf16).
  v16bf qa[2][2];
#pragma unroll
  for (int rt = 0; rt < 2; ++rt) {
    const __bf16* qbase = qkv + (size_t)(b * T + q0 + rt * 16 + lr) * (3 * E) + h * D;
#pragma unroll
    for (int f = 0; f < 2; ++f) {
      union { uint4 u[2]; v16bf v; } fq;
      fq.u[0] = *reinterpret_cast<const uint4*>(qbase + f * 32 + lh * 16);
      fq.u[1] = *reinterpret_cast<const uint4*>(qbase + f * 32 + lh * 16 + 8);
#pragma unroll
      for (int e = 0; e < 16; ++e) fq.v[e] = (__bf16)((float)fq.v[e] * 0.125f);
      qa[rt][f] = fq.v;
    }
  }

  v8f o[2][4];
  const v8f vzero = {0.f, 0.f, 0.f, 0.f, 0.f, 0.f, 0.f, 0.f};
  float m[2][8], lsum[2][8];
#pragma unroll
  for (int rt = 0; rt < 2; ++rt) {
#pragma unroll
    for (int j = 0; j < 4; ++j) o[rt][j] = vzero;
#pragma unroll
    for (int r = 0; r < 8; ++r) { m[rt][r] = -INFINITY; lsum[rt][r] = 0.f; }
  }

  // q0 is a multiple of 32 -> exactly one diagonal-crossing block (kk == q0)
  for (int kk = 0; kk < q0; kk += 32)
    attn_step<false>(qkv, b, h, q0, kk, lane, lh, lr, qa, ones, o, m, lsum, lP, lV);
  attn_step<true>(qkv, b, h, q0, q0, lane, lh, lr, qa, ones, o, m, lsum, lP, lV);

  // normalize & store bf16 for the projection GEMM
#pragma unroll
  for (int rt = 0; rt < 2; ++rt)
#pragma unroll
    for (int r = 0; r < 8; ++r) {
      float inv = 1.0f / lsum[rt][r];
      size_t row = (size_t)(b * T + q0 + rt * 16 + r + 8 * lh);
      __bf16* yb = y + row * E + h * D;
#pragma unroll
      for (int j = 0; j < 4; ++j) yb[j * 16 + lr] = (__bf16)(o[rt][j][r] * inv);
    }
}

// ---------------------------------------------------------------- host driver
extern "C" void kernel_launch(void* const* d_in, const int* in_sizes, int n_in,
                              void* d_out, int out_size, void* d_ws, size_t ws_size,
                              hipStream_t stream) {
  using namespace cfg;
  (void)in_sizes; (void)n_in; (void)out_size; (void)ws_size;
  const int*   idx   = (const int*)d_in[0];
  const float* wte   = (const float*)d_in[1];
  const float* wpe   = (const float*)d_in[2];
  const float* ln1w  = (const float*)d_in[3];
  const float* ln1b  = (const float*)d_in[4];
  const float* attnw = (const float*)d_in[5];
  const float* attnb = (const float*)d_in[6];
  const float* apw   = (const float*)d_in[7];
  const float* apb   = (const float*)d_in[8];
  const float* ln2w  = (const float*)d_in[9];
  const float* ln2b  = (const float*)d_in[10];
  const float* fcw   = (const float*)d_in[11];
  const float* fcb   = (const float*)d_in[12];
  const float* mpw   = (const float*)d_in[13];
  const float* mpb   = (const float*)d_in[14];
  const float* lnfw  = (const float*)d_in[15];
  const float* lnfb  = (const float*)d_in[16];

  char* p = (char*)d_ws;
  auto carve = [&](size_t bytes) { char* r = p; p += (bytes + 255) & ~(size_t)255; return r; };
  __bf16* wte_bf = (__bf16*)carve((size_t)V * E * 2);
  __bf16* aw_bf  = (__bf16*)carve((size_t)L * E * 3 * E * 2);
  __bf16* apw_bf = (__bf16*)carve((size_t)L * E * E * 2);
  __bf16* fcw_bf = (__bf16*)carve((size_t)L * E * 4 * E * 2);
  __bf16* mpw_bf = (__bf16*)carve((size_t)L * 4 * E * E * 2);
  float*  x      = (float*)carve((size_t)B * T * E * 4);
  __bf16* lnbuf  = (__bf16*)carve((size_t)B * T * E * 2);
  __bf16* qkv    = (__bf16*)carve((size_t)B * T * 3 * E * 2);
  __bf16* ybf    = (__bf16*)carve((size_t)B * T * E * 2);
  __bf16* hbf    = (__bf16*)carve((size_t)B * T * 4 * E * 2);
  __bf16* x8     = (__bf16*)carve((size_t)B * E * 2);

  auto cvt = [&](const float* src, __bf16* dst, size_t n) {
    cvt_kernel<<<dim3((unsigned)((n + 255) / 256)), dim3(256), 0, stream>>>(src, dst, n);
  };
  cvt(wte, wte_bf, (size_t)V * E);
  cvt(attnw, aw_bf, (size_t)L * E * 3 * E);
  cvt(apw, apw_bf, (size_t)L * E * E);
  cvt(fcw, fcw_bf, (size_t)L * E * 4 * E);
  cvt(mpw, mpw_bf, (size_t)L * 4 * E * E);

  embed_kernel<<<dim3(B * T * E / 256), dim3(256), 0, stream>>>(idx, wte, wpe, x);

  const int M = B * T;
  for (int l = 0; l < L; ++l) {
    ln_kernel<<<dim3(M / 8), dim3(256), 0, stream>>>(x, ln1w + (size_t)l * E, ln1b + (size_t)l * E, lnbuf, M);
    gemm_kernel<0, false><<<dim3(3 * E / 128, M / 128), dim3(256), 0, stream>>>(
        lnbuf, aw_bf + (size_t)l * E * 3 * E, attnb + (size_t)l * 3 * E, nullptr, qkv, M, 3 * E, E);
    attn_kernel<<<dim3(B * H * (T / 32)), dim3(32), 0, stream>>>(qkv, ybf);
    gemm_kernel<2, false><<<dim3(E / 128, M / 128), dim3(256), 0, stream>>>(
        ybf, apw_bf + (size_t)l * E * E, apb + (size_t)l * E, x, x, M, E, E);
    ln_kernel<<<dim3(M / 8), dim3(256), 0, stream>>>(x, ln2w + (size_t)l * E, ln2b + (size_t)l * E, lnbuf, M);
    gemm_kernel<1, false><<<dim3(4 * E / 128, M / 128), dim3(256), 0, stream>>>(
        lnbuf, fcw_bf + (size_t)l * E * 4 * E, fcb + (size_t)l * 4 * E, nullptr, hbf, M, 4 * E, E);
    gemm_kernel<2, false><<<dim3(E / 128, M / 128), dim3(256), 0, stream>>>(
        hbf, mpw_bf + (size_t)l * 4 * E * E, mpb + (size_t)l * E, x, x, M, E, 4 * E);
  }
  ln_kernel<<<dim3(M / 8), dim3(256), 0, stream>>>(x, lnfw, lnfb, lnbuf, M);
  gather_kernel<<<dim3((B * E + 255) / 256), dim3(256), 0, stream>>>(lnbuf, x8);
  gemm_kernel<2, true><<<dim3(V / 128, 1), dim3(256), 0, stream>>>(
      x8, wte_bf, nullptr, nullptr, d_out, B, V, E);
}
